// SubgraphMamba_32074815767060
// MI455X (gfx1250) — matrix operand order repaired and checked by reading, not compiled, not run on hardware
//
#include <hip/hip_runtime.h>
#include <hip/hip_bf16.h>

typedef __attribute__((ext_vector_type(16))) _Float16 v16h;
typedef __attribute__((ext_vector_type(8)))  _Float16 v8h;
typedef __attribute__((ext_vector_type(4)))  _Float16 v4h;
typedef __attribute__((ext_vector_type(8)))  float    v8f;

#define NNODES 50000
#define FDIM   128
#define HIDD   256
#define EEDGE  800000
#define BB     32
#define LL     1024
#define DMM    256
#define DNN    512     // d_inner
#define DSS    16      // d_state
#define DTRK   16      // dt_rank
#define NCLS   10
#define BLTOT  (BB*LL) // 32768

// ---------------- GCN degree / normalization ----------------
__global__ void k_deg_init(float* deg) {
    int i = blockIdx.x * 256 + threadIdx.x;
    if (i < NNODES) deg[i] = 1.0f;                 // self-loop contributes 1
}
__global__ void k_deg_count(const int* ei, float* deg) {
    int e = blockIdx.x * 256 + threadIdx.x;
    if (e < EEDGE) atomicAdd(&deg[ei[EEDGE + e]], 1.0f);   // dst = edge_index[1]
}
__global__ void k_dinv(float* deg) {
    int i = blockIdx.x * 256 + threadIdx.x;
    if (i < NNODES) deg[i] = rsqrtf(fmaxf(deg[i], 1.0f));
}

// ---------------- pad fc1 weights/bias to 256 cols ----------------
__global__ void k_pad_fc1(const float* w, const float* b, float* wpad, float* bpad) {
    int idx = blockIdx.x * 256 + threadIdx.x;      // 65536 threads
    int k = idx >> 8, n = idx & 255;
    wpad[idx] = (n < DMM - 1) ? w[k * (DMM - 1) + n] : 0.f;
    if (idx < DMM) bpad[idx] = (idx < DMM - 1) ? b[idx] : 0.f;
}

// ---------------- LDS-staged WMMA f16->f32 GEMM ----------------
// C[M,N] = A[M,K] @ Bw[K,N] (+ bias).  N % 128 == 0, K % 32 == 0, M arbitrary (>=16).
// Block = 256 threads = 8 waves computing a 32x128 strip; 2 WMMAs per wave per k-chunk.
__global__ void k_gemm_lds(const float* __restrict__ A, const float* __restrict__ Bw,
                           const float* __restrict__ bias, float* __restrict__ C,
                           int M, int N, int K) {
    __shared__ _Float16 As[32 * 32];               // [row][k]  k contiguous
    __shared__ _Float16 Bs[128 * 32];              // [col][k]  k contiguous
    const int tid  = threadIdx.x;
    const int lane = tid & 31;
    const int wv   = tid >> 5;
    const int half = lane >> 4;
    const int tm   = blockIdx.y;                   // 32-row tile
    const int cg   = blockIdx.x * 128;             // 128-col group base
    const int col  = cg + wv * 16 + (lane & 15);   // this lane's output column

    // fixed staging assignment: A -> one float4 ; B -> one 4x4 block
    const int ar  = tid >> 3;                      // A row 0..31
    const int ak  = (tid & 7) << 2;                // A k offset 0..28
    const int bkb = (tid >> 5) << 2;               // B k-block 0..28
    const int bcb = (tid & 31) << 2;               // B col-block 0..124

    v8f acc0 = {0.f,0.f,0.f,0.f,0.f,0.f,0.f,0.f};
    v8f acc1 = {0.f,0.f,0.f,0.f,0.f,0.f,0.f,0.f};
    for (int k0 = 0; k0 < K; k0 += 32) {
        // ---- stage A tile (32x32): one b128 load + one b64 LDS store per thread
        int grow = tm * 32 + ar;  grow = (grow < M) ? grow : (M - 1);   // clamp, no divergence
        float4 av = *(const float4*)&A[(size_t)grow * K + k0 + ak];
        v4h ap = { (_Float16)av.x, (_Float16)av.y, (_Float16)av.z, (_Float16)av.w };
        *(v4h*)&As[ar * 32 + ak] = ap;
        // ---- stage B panel (32x128): 4 coalesced b128 loads, 4x4 register transpose,
        //      4 b64 LDS stores (K-contiguous per column)
        const float* bp0 = &Bw[(size_t)(k0 + bkb) * N + cg + bcb];
        float4 b0 = *(const float4*)(bp0);
        float4 b1 = *(const float4*)(bp0 + N);
        float4 b2 = *(const float4*)(bp0 + 2 * (size_t)N);
        float4 b3 = *(const float4*)(bp0 + 3 * (size_t)N);
        v4h c0 = { (_Float16)b0.x, (_Float16)b1.x, (_Float16)b2.x, (_Float16)b3.x };
        v4h c1 = { (_Float16)b0.y, (_Float16)b1.y, (_Float16)b2.y, (_Float16)b3.y };
        v4h c2 = { (_Float16)b0.z, (_Float16)b1.z, (_Float16)b2.z, (_Float16)b3.z };
        v4h c3 = { (_Float16)b0.w, (_Float16)b1.w, (_Float16)b2.w, (_Float16)b3.w };
        *(v4h*)&Bs[(bcb + 0) * 32 + bkb] = c0;
        *(v4h*)&Bs[(bcb + 1) * 32 + bkb] = c1;
        *(v4h*)&Bs[(bcb + 2) * 32 + bkb] = c2;
        *(v4h*)&Bs[(bcb + 3) * 32 + bkb] = c3;
        // branch-free prefetch of next weight panel (address clamped in-range)
        int kp = (k0 + 32 < K) ? (k0 + 32) : k0;
        __builtin_prefetch(&Bw[(size_t)(kp + bkb) * N + cg + bcb], 0, 3);
        __syncthreads();

        // ---- compute: two 16x16 tiles share one B operand
        int arow = lane & 15;
        v8h alo0 = *(const v8h*)&As[arow * 32 + half * 8];
        v8h ahi0 = *(const v8h*)&As[arow * 32 + 16 + half * 8];
        v8h alo1 = *(const v8h*)&As[(16 + arow) * 32 + half * 8];
        v8h ahi1 = *(const v8h*)&As[(16 + arow) * 32 + 16 + half * 8];
        v16h a0 = __builtin_shufflevector(alo0, ahi0, 0,1,2,3,4,5,6,7,8,9,10,11,12,13,14,15);
        v16h a1 = __builtin_shufflevector(alo1, ahi1, 0,1,2,3,4,5,6,7,8,9,10,11,12,13,14,15);
        v8h blo = *(const v8h*)&Bs[(wv * 16 + (lane & 15)) * 32 + half * 8];
        v8h bhi = *(const v8h*)&Bs[(wv * 16 + (lane & 15)) * 32 + 16 + half * 8];
        v16h b  = __builtin_shufflevector(blo, bhi, 0,1,2,3,4,5,6,7,8,9,10,11,12,13,14,15);

        acc0 = __builtin_amdgcn_wmma_f32_16x16x32_f16(false, a0, false, b,
                                                      (short)0, acc0, false, false);
        acc1 = __builtin_amdgcn_wmma_f32_16x16x32_f16(false, a1, false, b,
                                                      (short)0, acc1, false, false);
        __syncthreads();
    }
    float bb = (bias != nullptr) ? bias[col] : 0.f;
    #pragma unroll
    for (int i = 0; i < 8; ++i) {                  // C/D: vgpr i -> m = i + half*8
        int m0 = tm * 32 + half * 8 + i;
        int m1 = m0 + 16;
        if (m0 < M) C[(size_t)m0 * N + col] = acc0[i] + bb;
        if (m1 < M) C[(size_t)m1 * N + col] = acc1[i] + bb;
    }
}

// ---------------- simple per-wave WMMA GEMM (small N, here N=48) ----------------
__global__ void k_gemm_small(const float* __restrict__ A, const float* __restrict__ Bw,
                             float* __restrict__ C, int M, int N, int K, int tilesTotal) {
    int lane = threadIdx.x & 31;
    int wv   = threadIdx.x >> 5;
    int tile = blockIdx.x * (blockDim.x >> 5) + wv;
    if (tile >= tilesTotal) return;                // wave-uniform
    int tilesN = N >> 4;
    int tm = tile / tilesN, tn = tile % tilesN;
    int half = lane >> 4;
    int row  = tm * 16 + (lane & 15);
    int col  = tn * 16 + (lane & 15);

    v8f acc = {0.f,0.f,0.f,0.f,0.f,0.f,0.f,0.f};
    for (int k0 = 0; k0 < K; k0 += 32) {
        v16h a, b;
        #pragma unroll
        for (int e = 0; e < 16; ++e) {
            int v = e >> 1, pos = e & 1;
            int kk = k0 + ((v < 4) ? 0 : 16) + half * 8 + ((v & 3) << 1) + pos;
            a[e] = (_Float16)A[(size_t)row * K + kk];
        }
        #pragma unroll
        for (int e = 0; e < 16; ++e)
            b[e] = (_Float16)Bw[(size_t)(k0 + ((e < 8) ? 0 : 16) + half * 8 + (e & 7)) * N + col];
        acc = __builtin_amdgcn_wmma_f32_16x16x32_f16(false, a, false, b,
                                                     (short)0, acc, false, false);
    }
    #pragma unroll
    for (int i = 0; i < 8; ++i) {
        int m = tm * 16 + half * 8 + i;
        C[(size_t)m * N + col] = acc[i];
    }
}

// ---------------- GCN aggregate ----------------
__global__ void k_h_init(const float* xw, const float* dinv, const float* b_gcn, float* h) {
    int i = blockIdx.x, c = threadIdx.x;           // self-loop term + bias
    float dn = dinv[i];
    h[(size_t)i * HIDD + c] = b_gcn[c] + xw[(size_t)i * HIDD + c] * dn * dn;
}
__global__ void k_edge_scatter(const int* ei, const float* xw, const float* dinv, float* h) {
    int e = blockIdx.x, c = threadIdx.x;
    int s = ei[e], d = ei[EEDGE + e];
    float nrm = dinv[s] * dinv[d];
    atomicAdd(&h[(size_t)d * HIDD + c], xw[(size_t)s * HIDD + c] * nrm);
}

// ---------------- gather + inclusion concat ----------------
__global__ void k_gather_x(const float* emb, const int* seq, const float* incl, float* x) {
    int r = blockIdx.x;                            // r = b*L + l
    int j = threadIdx.x;
    int node = seq[r];
    float v = (j < DMM - 1) ? emb[(size_t)node * DMM + j] : incl[r];
    x[(size_t)r * DMM + j] = v;
}

// ---------------- causal depthwise conv (k=4) + SiLU ----------------
__global__ void k_conv_silu(const float* xz, const float* cw, const float* cb, float* xm) {
    long idx = (long)blockIdx.x * 256 + threadIdx.x;
    if (idx >= (long)BLTOT * DNN) return;
    int c = idx % DNN;
    long r = idx / DNN;                            // r = b*L + l
    int l = r % LL;
    long rb = r - l;                               // row of l=0 in this batch
    float acc = cb[c];
    #pragma unroll
    for (int i = 0; i < 4; ++i) {
        int ls = l + i - 3;
        float xv = (ls >= 0) ? xz[((size_t)(rb + ls)) * (2 * DNN) + c] : 0.f;
        acc += xv * cw[i * DNN + c];
    }
    xm[idx] = acc / (1.f + __expf(-acc));          // silu
}

// ---------------- delta = softplus(dt @ dt_proj_w + b) ----------------
__global__ void k_delta(const float* dbl, const float* w, const float* b, float* delta) {
    long idx = (long)blockIdx.x * 256 + threadIdx.x;
    if (idx >= (long)BLTOT * DNN) return;
    int d = idx % DNN;
    long r = idx / DNN;
    float acc = b[d];
    #pragma unroll
    for (int t = 0; t < DTRK; ++t)
        acc += dbl[(size_t)r * 48 + t] * w[t * DNN + d];
    delta[idx] = (acc > 20.f) ? acc : log1pf(__expf(acc));
}

// ---------------- selective scan: one thread per (b,d), 16-state in regs ----------------
__global__ void k_scan(const float* __restrict__ xz, const float* __restrict__ xm,
                       const float* __restrict__ dbl, const float* __restrict__ delta,
                       const float* __restrict__ A_log, const float* __restrict__ Dp,
                       float* __restrict__ y) {
    int b = blockIdx.x, d = threadIdx.x;           // blockDim = DNN
    __shared__ float sB[DSS], sC[DSS];
    float Ar[DSS];
    #pragma unroll
    for (int s = 0; s < DSS; ++s) Ar[s] = -__expf(A_log[(size_t)d * DSS + s]);
    float hst[DSS];
    #pragma unroll
    for (int s = 0; s < DSS; ++s) hst[s] = 0.f;
    float Dd = Dp[d];

    for (int l = 0; l < LL; ++l) {
        long r = (long)b * LL + l;
        if (d < DSS)            sB[d]        = dbl[(size_t)r * 48 + DTRK + d];
        else if (d < 2 * DSS)   sC[d - DSS]  = dbl[(size_t)r * 48 + DTRK + DSS + (d - DSS)];
        __syncthreads();
        float dt = delta[(size_t)r * DNN + d];
        float xt = xm[(size_t)r * DNN + d];
        float zt = xz[(size_t)r * (2 * DNN) + DNN + d];
        float dx = dt * xt;
        float acc = 0.f;
        #pragma unroll
        for (int s = 0; s < DSS; ++s) {
            hst[s] = hst[s] * __expf(dt * Ar[s]) + dx * sB[s];
            acc += hst[s] * sC[s];
        }
        float yy = acc + xt * Dd;
        float sz = zt / (1.f + __expf(-zt));
        y[(size_t)r * DNN + d] = yy * sz;
        __syncthreads();
    }
}

// ---------------- masked mean pool over cat([subg, subg0]) ----------------
__global__ void k_pool(const float* subg, const float* x, const float* incl, float* pooled) {
    int b = blockIdx.x, j = threadIdx.x;
    if (j >= 2 * DMM - 1) return;                  // 511 real cols
    float sum = 0.f, msum = 0.f;
    for (int l = 0; l < LL; ++l) {
        long r = (long)b * LL + l;
        float m = incl[r];
        float v = (j < DMM) ? subg[(size_t)r * DMM + j]
                            : x[(size_t)r * DMM + (j - DMM)];   // subg0 = x[:, :255]
        sum += v * m; msum += m;
    }
    pooled[b * 512 + j] = sum / (msum + 1e-5f);
}

__global__ void k_fc2(const float* pooled, const float* w, const float* b, float* out) {
    int t = threadIdx.x;
    if (t >= BB * NCLS) return;
    int bi = t / NCLS, c = t % NCLS;
    float acc = b[c];
    for (int j = 0; j < 2 * DMM - 1; ++j)
        acc += pooled[bi * 512 + j] * w[j * NCLS + c];
    out[t] = acc;
}

extern "C" void kernel_launch(void* const* d_in, const int* in_sizes, int n_in,
                              void* d_out, int out_size, void* d_ws, size_t ws_size,
                              hipStream_t stream) {
    (void)in_sizes; (void)n_in; (void)out_size; (void)ws_size;
    const float* embeddings = (const float*)d_in[0];
    const float* inclusion  = (const float*)d_in[1];
    const float* W_gcn      = (const float*)d_in[2];
    const float* b_gcn      = (const float*)d_in[3];
    const float* W_fc1      = (const float*)d_in[4];
    const float* b_fc1      = (const float*)d_in[5];
    const float* in_proj_w  = (const float*)d_in[6];
    const float* conv_w     = (const float*)d_in[7];
    const float* conv_b     = (const float*)d_in[8];
    const float* x_proj_w   = (const float*)d_in[9];
    const float* dt_proj_w  = (const float*)d_in[10];
    const float* dt_proj_b  = (const float*)d_in[11];
    const float* A_log      = (const float*)d_in[12];
    const float* Dp         = (const float*)d_in[13];
    const float* out_proj_w = (const float*)d_in[14];
    const float* W_fc2      = (const float*)d_in[15];
    const float* b_fc2      = (const float*)d_in[16];
    const int*   edge_index = (const int*)d_in[17];
    const int*   sequence   = (const int*)d_in[18];
    float* out = (float*)d_out;

    // workspace partition (floats)
    float* ws = (float*)d_ws;
    size_t off = 0;
    float* dinv  = ws + off; off += 50048;                       // deg -> dinv in place
    float* xw    = ws + off; off += (size_t)NNODES * HIDD;       // 12.8M (reused for subg)
    float* hbuf  = ws + off; off += (size_t)NNODES * HIDD;       // 12.8M
    float* emb   = ws + off; off += (size_t)NNODES * DMM;        // 12.8M (padded col 255)
    float* x     = ws + off; off += (size_t)BLTOT * DMM;         // 8.39M
    float* xz    = ws + off; off += (size_t)BLTOT * 2 * DNN;     // 33.6M
    float* xm    = ws + off; off += (size_t)BLTOT * DNN;         // 16.8M
    float* dbl   = ws + off; off += (size_t)BLTOT * 48;          // 1.57M
    float* delta = ws + off; off += (size_t)BLTOT * DNN;         // 16.8M
    float* ybuf  = ws + off; off += (size_t)BLTOT * DNN;         // 16.8M
    float* wpad  = ws + off; off += (size_t)HIDD * DMM;          // 65536 (padded W_fc1)
    float* bpad  = ws + off; off += DMM;                         // 256   (padded b_fc1)
    float* subg  = xw;                                           // xw dead after fc1
    float* pooled= dinv;                                         // dinv dead after scatter

    // --- GCN ---
    k_deg_init <<<(NNODES + 255) / 256, 256, 0, stream>>>(dinv);
    k_deg_count<<<(EEDGE + 255) / 256, 256, 0, stream>>>(edge_index, dinv);
    k_dinv     <<<(NNODES + 255) / 256, 256, 0, stream>>>(dinv);
    k_pad_fc1  <<<(HIDD * DMM) / 256, 256, 0, stream>>>(W_fc1, b_fc1, wpad, bpad);

    // xw = embeddings @ W_gcn   [50000,128]x[128,256]
    k_gemm_lds<<<dim3(HIDD / 128, (NNODES + 31) / 32), 256, 0, stream>>>(
        embeddings, W_gcn, nullptr, xw, NNODES, HIDD, FDIM);
    k_h_init      <<<NNODES, HIDD, 0, stream>>>(xw, dinv, b_gcn, hbuf);
    k_edge_scatter<<<EEDGE, HIDD, 0, stream>>>(edge_index, xw, dinv, hbuf);

    // emb = h @ W_fc1 + b_fc1   [50000,256]x[256,256-padded]
    k_gemm_lds<<<dim3(DMM / 128, (NNODES + 31) / 32), 256, 0, stream>>>(
        hbuf, wpad, bpad, emb, NNODES, DMM, HIDD);

    // --- gather + Mamba ---
    k_gather_x<<<BLTOT, DMM, 0, stream>>>(emb, sequence, inclusion, x);

    // xz = x @ in_proj_w        [32768,256]x[256,1024]
    k_gemm_lds<<<dim3((2 * DNN) / 128, BLTOT / 32), 256, 0, stream>>>(
        x, in_proj_w, nullptr, xz, BLTOT, 2 * DNN, DMM);

    k_conv_silu<<<(int)(((long)BLTOT * DNN + 255) / 256), 256, 0, stream>>>(xz, conv_w, conv_b, xm);

    // dbl = xm @ x_proj_w       [32768,512]x[512,48]  (small-N path)
    { int tiles = (BLTOT / 16) * (48 / 16);
      k_gemm_small<<<(tiles + 7) / 8, 256, 0, stream>>>(xm, x_proj_w, dbl,
                                                        BLTOT, 48, DNN, tiles); }

    k_delta<<<(int)(((long)BLTOT * DNN + 255) / 256), 256, 0, stream>>>(dbl, dt_proj_w, dt_proj_b, delta);

    k_scan<<<BB, DNN, 0, stream>>>(xz, xm, dbl, delta, A_log, Dp, ybuf);

    // subg = y @ out_proj_w     [32768,512]x[512,256]
    k_gemm_lds<<<dim3(DMM / 128, BLTOT / 32), 256, 0, stream>>>(
        ybuf, out_proj_w, nullptr, subg, BLTOT, DMM, DNN);

    // --- pool + classifier ---
    k_pool<<<BB, 512, 0, stream>>>(subg, x, inclusion, pooled);
    k_fc2 <<<1, 320, 0, stream>>>(pooled, W_fc2, b_fc2, out);
}
